// GAT_1211180778302
// MI455X (gfx1250) — matrix operand order repaired and checked by reading, not compiled, not run on hardware
//
#include <hip/hip_runtime.h>
#include <hip/hip_bf16.h>

typedef __attribute__((ext_vector_type(16))) _Float16 v16h;
typedef __attribute__((ext_vector_type(8)))  _Float16 v8h;
typedef __attribute__((ext_vector_type(8)))  float    v8f;

constexpr int Nn    = 50000;
constexpr int Ee    = 800000;
constexpr int Gg    = 64;
constexpr int KIN   = 128;
constexpr int FDIM  = 256;   // HEADS*HID
constexpr int HEADS = 8;
constexpr int HID   = 32;
constexpr int NCLS  = 10;
constexpr int NPAD  = 50016; // round N up to multiple of 32 (GEMM block = 32 rows)
constexpr float SLOPE = 0.2f;

// ---- monotone float<->uint key for atomic max on floats ----
__device__ __forceinline__ unsigned fkey(float f) {
  unsigned u = __float_as_uint(f);
  return (u & 0x80000000u) ? ~u : (u | 0x80000000u);
}
__device__ __forceinline__ float funkey(unsigned u) {
  return (u & 0x80000000u) ? __uint_as_float(u & 0x7fffffffu) : __uint_as_float(~u);
}

// ---------------- utility kernels ----------------
__global__ void zero_f32(float* __restrict__ p, int n) {
  int i = blockIdx.x * blockDim.x + threadIdx.x;
  if (i < n) p[i] = 0.f;
}

// fp32 node features -> f16, zero-fill pad rows. Layer-1 input (K = 128).
__global__ void cvt_in(const float* __restrict__ h, _Float16* __restrict__ x) {
  int i = blockIdx.x * blockDim.x + threadIdx.x;
  if (i >= NPAD * KIN) return;
  int n = i >> 7;
  x[i] = (n < Nn) ? (_Float16)h[i] : (_Float16)0.0f;
}

// W [K,256] row-major fp32 -> Wt [256][K] f16 (column becomes contiguous k-run)
__global__ void cvt_wt(const float* __restrict__ W, _Float16* __restrict__ Wt, int K) {
  int i = blockIdx.x * blockDim.x + threadIdx.x;
  if (i >= FDIM * K) return;
  int n = i / K, k = i - n * K;
  Wt[i] = (_Float16)W[k * FDIM + n];
}

// ---------------- WMMA GEMM: C[NPAD,256] = A[NPAD,K](f16) x Bt[256,K]^T(f16) ----------------
// Block = 8 waves = 32 rows x 256 cols. Wave (w): rows (w>>2)*16, cols (w&3)*64 (4 x 16x16 tiles).
// B tile (256 cols x 32 k = 16KB) is staged into LDS once per block per k-step through the
// CDNA5 async-to-LDS datapath (GLOBAL_LOAD_ASYNC_TO_LDS_B128, ASYNCcnt), then read with ds_load.
__global__ __launch_bounds__(256) void gemm_wmma(
    const _Float16* __restrict__ A, const _Float16* __restrict__ Bt,
    float* __restrict__ C, int K)
{
  __shared__ __align__(16) _Float16 tileB[FDIM * 32];  // [col][k_local] : 64B per column

  const int wave = threadIdx.x >> 5;
  const int lane = threadIdx.x & 31;
  const int hi   = lane >> 4;
  const int l15  = lane & 15;
  const int rowBase = blockIdx.x * 32 + (wave >> 2) * 16;
  const int colBase = (wave & 3) * 64;

  // low 32 bits of the generic LDS address == byte offset within the LDS allocation
  const unsigned ldsDst = (unsigned)(unsigned long long)(const void*)(tileB) +
                          (unsigned)threadIdx.x * 64u;
  const unsigned long long gsrcBase =
      (unsigned long long)(const void*)(Bt + (size_t)threadIdx.x * K);

  v8f acc[4] = {};
  const _Float16* arow = A + (size_t)(rowBase + l15) * K + hi * 8;

  for (int kk = 0; kk < K; kk += 32) {
    // ---- stage B tile: thread t copies column t's 32 k-values (64B) via async B128 copies ----
    {
      unsigned long long g = gsrcBase + (unsigned)kk * 2u;  // halves -> bytes
#pragma unroll
      for (int q = 0; q < 4; ++q) {
        asm volatile("global_load_async_to_lds_b128 %0, %1, off"
                     :: "v"(ldsDst + q * 16u), "v"(g + q * 16u) : "memory");
      }
      asm volatile("s_wait_asynccnt 0x0" ::: "memory");
    }
    __syncthreads();  // tile visible to all waves

    if (kk + 32 < K) __builtin_prefetch(arow + kk + 32, 0, 1);
    // A fragment (16x32 f16): lanes 0-15 hold K 0..7 & 16..23 of row M=lane;
    // lanes 16-31 hold K 8..15 & 24..31 -> two contiguous b128 loads per lane.
    v8h a0 = *(const v8h*)(arow + kk);
    v8h a1 = *(const v8h*)(arow + kk + 16);
    v16h af = __builtin_shufflevector(a0, a1, 0,1,2,3,4,5,6,7,8,9,10,11,12,13,14,15);
#pragma unroll
    for (int c = 0; c < 4; ++c) {
      // B fragment (32x16 f16): lane n holds column n, k_local = 8*hi + j (j<8), 16 + 8*hi + (j-8) (j>=8)
      const _Float16* bp = &tileB[(colBase + c * 16 + l15) * 32 + hi * 8];
      v8h b0 = *(const v8h*)bp;          // ds_load_b128
      v8h b1 = *(const v8h*)(bp + 16);   // ds_load_b128
      v16h bf = __builtin_shufflevector(b0, b1, 0,1,2,3,4,5,6,7,8,9,10,11,12,13,14,15);
      acc[c] = __builtin_amdgcn_wmma_f32_16x16x32_f16(
          /*neg_a=*/false, af, /*neg_b=*/false, bf,
          /*c_mod=*/(short)0, acc[c], /*reuse_a=*/false, /*reuse_b=*/false);
    }
    __syncthreads();  // all waves done reading before next stage overwrites
  }
  // C/D layout: VGPR v, lanes 0-15 -> M=v, N=lane; lanes 16-31 -> M=8+v, N=lane-16
#pragma unroll
  for (int c = 0; c < 4; ++c) {
    float* crow = C + (size_t)(rowBase + hi * 8) * FDIM + colBase + c * 16 + l15;
#pragma unroll
    for (int v = 0; v < 8; ++v) crow[(size_t)v * FDIM] = acc[c][v];
  }
}

// ---------------- per-node attention scores ----------------
__global__ void attn_lr(const float* __restrict__ feat,
                        const float* __restrict__ al, const float* __restrict__ ar,
                        float* __restrict__ el, float* __restrict__ er) {
  int t = blockIdx.x * blockDim.x + threadIdx.x;
  if (t >= Nn * HEADS) return;
  int n = t >> 3, h = t & 7;
  const float* f = feat + (size_t)n * FDIM + h * HID;
  const float* a = al + h * HID;
  const float* b = ar + h * HID;
  float sl = 0.f, sr = 0.f;
#pragma unroll
  for (int d = 0; d < HID; ++d) { sl += f[d] * a[d]; sr += f[d] * b[d]; }
  el[t] = sl;
  er[t] = sr;
}

// ---------------- edge passes ----------------
__device__ __forceinline__ float edge_e(const float* el, const float* er,
                                        int s, int d, int h) {
  float x = el[s * HEADS + h] + er[d * HEADS + h];
  return x > 0.f ? x : SLOPE * x;
}

__global__ void edge_max(const int* __restrict__ src, const int* __restrict__ dst,
                         const float* __restrict__ el, const float* __restrict__ er,
                         unsigned* __restrict__ mkey) {
  int t = blockIdx.x * blockDim.x + threadIdx.x;
  if (t >= Ee * HEADS) return;
  int e = t >> 3, h = t & 7;
  int s = src[e], d = dst[e];
  atomicMax(&mkey[d * HEADS + h], fkey(edge_e(el, er, s, d, h)));
}

__global__ void edge_den(const int* __restrict__ src, const int* __restrict__ dst,
                         const float* __restrict__ el, const float* __restrict__ er,
                         const unsigned* __restrict__ mkey, float* __restrict__ den) {
  int t = blockIdx.x * blockDim.x + threadIdx.x;
  if (t >= Ee * HEADS) return;
  int e = t >> 3, h = t & 7;
  int s = src[e], d = dst[e];
  float x = edge_e(el, er, s, d, h);
  float ee = __expf(x - funkey(mkey[d * HEADS + h]));
  atomicAdd(&den[d * HEADS + h], ee);
}

__global__ void edge_msg(const int* __restrict__ src, const int* __restrict__ dst,
                         const float* __restrict__ el, const float* __restrict__ er,
                         const unsigned* __restrict__ mkey, const float* __restrict__ den,
                         const float* __restrict__ feat, float* __restrict__ agg) {
  int t = blockIdx.x * blockDim.x + threadIdx.x;
  if (t >= Ee * HEADS) return;
  int e = t >> 3, h = t & 7;
  int s = src[e], d = dst[e];
  float x = edge_e(el, er, s, d, h);
  float ee = __expf(x - funkey(mkey[d * HEADS + h]));
  float alpha = ee / den[d * HEADS + h];
  const float4* fs = (const float4*)(feat + (size_t)s * FDIM + h * HID);
  float* ag = agg + (size_t)d * FDIM + h * HID;
#pragma unroll
  for (int q = 0; q < HID / 4; ++q) {
    float4 v = fs[q];
    atomicAdd(ag + q * 4 + 0, v.x * alpha);
    atomicAdd(ag + q * 4 + 1, v.y * alpha);
    atomicAdd(ag + q * 4 + 2, v.z * alpha);
    atomicAdd(ag + q * 4 + 3, v.w * alpha);
  }
}

// ---------------- fused bias + relu + f16 convert (layer output -> next GEMM input) ----------------
__global__ void bias_relu_cvt(const float* __restrict__ agg, const float* __restrict__ b,
                              _Float16* __restrict__ x) {
  int i = blockIdx.x * blockDim.x + threadIdx.x;
  if (i >= NPAD * FDIM) return;
  int n = i >> 8, c = i & 255;
  float v = 0.f;
  if (n < Nn) v = fmaxf(agg[i] + b[c], 0.f);
  x[i] = (_Float16)v;
}

// ---------------- per-graph mean pooling (sum + count via atomics) ----------------
__global__ void pool_nodes(const float* __restrict__ agg, const float* __restrict__ b,
                           const int* __restrict__ gid, float* __restrict__ pooled,
                           float* __restrict__ cnt) {
  int i = blockIdx.x * blockDim.x + threadIdx.x;
  if (i >= Nn * FDIM) return;
  int n = i >> 8, c = i & 255;
  float v = fmaxf(agg[i] + b[c], 0.f);
  int g = gid[n];
  atomicAdd(&pooled[g * FDIM + c], v);
  if (c == 0) atomicAdd(&cnt[g], 1.f);
}

// ---------------- MLP head, single block (G=64 tiny) ----------------
__global__ __launch_bounds__(256) void mlp_head(
    const float* __restrict__ pooled, const float* __restrict__ cnt,
    const float* __restrict__ Wc1, const float* __restrict__ bc1,
    const float* __restrict__ Wc2, const float* __restrict__ bc2,
    const float* __restrict__ Wc3, const float* __restrict__ bc3,
    float* __restrict__ out) {
  __shared__ float o1[Gg * HID];
  __shared__ float o2[Gg * HID];
  int t = threadIdx.x;
  for (int i = t; i < Gg * HID; i += 256) {
    int g = i >> 5, j = i & 31;
    float inv = 1.f / fmaxf(cnt[g], 1.f);
    float s = 0.f;
    for (int k = 0; k < FDIM; ++k) s += pooled[g * FDIM + k] * Wc1[k * HID + j];
    o1[i] = fmaxf(s * inv + bc1[j], 0.f);
  }
  __syncthreads();
  for (int i = t; i < Gg * HID; i += 256) {
    int g = i >> 5, j = i & 31;
    float s = bc2[j];
    for (int k = 0; k < HID; ++k) s += o1[g * HID + k] * Wc2[k * HID + j];
    o2[i] = fmaxf(s, 0.f);
  }
  __syncthreads();
  for (int i = t; i < Gg * NCLS; i += 256) {
    int g = i / NCLS, j = i - g * NCLS;
    float s = bc3[j];
    for (int k = 0; k < HID; ++k) s += o2[g * HID + k] * Wc3[k * NCLS + j];
    out[i] = s;
  }
}

static inline int cdiv(long long a, long long b) { return (int)((a + b - 1) / b); }

extern "C" void kernel_launch(void* const* d_in, const int* in_sizes, int n_in,
                              void* d_out, int out_size, void* d_ws, size_t ws_size,
                              hipStream_t stream) {
  (void)in_sizes; (void)n_in; (void)out_size; (void)ws_size;
  const float* h   = (const float*)d_in[0];
  const int*   src = (const int*)d_in[1];
  const int*   dst = (const int*)d_in[2];
  const int*   gid = (const int*)d_in[3];
  const float* W1  = (const float*)d_in[4];
  const float* al1 = (const float*)d_in[5];
  const float* ar1 = (const float*)d_in[6];
  const float* b1  = (const float*)d_in[7];
  const float* W2  = (const float*)d_in[8];
  const float* al2 = (const float*)d_in[9];
  const float* ar2 = (const float*)d_in[10];
  const float* b2  = (const float*)d_in[11];
  const float* Wc1 = (const float*)d_in[12];
  const float* bc1 = (const float*)d_in[13];
  const float* Wc2 = (const float*)d_in[14];
  const float* bc2 = (const float*)d_in[15];
  const float* Wc3 = (const float*)d_in[16];
  const float* bc3 = (const float*)d_in[17];

  // workspace layout (all 16B-aligned)
  char* ws = (char*)d_ws;
  size_t o = 0;
  float*    feat   = (float*)(ws + o);    o += (size_t)NPAD * FDIM * 4;
  _Float16* x16    = (_Float16*)(ws + o); o += (size_t)NPAD * FDIM * 2;
  _Float16* wt16   = (_Float16*)(ws + o); o += (size_t)FDIM * FDIM * 2;
  float*    el     = (float*)(ws + o);    o += (size_t)Nn * HEADS * 4;
  float*    er     = (float*)(ws + o);    o += (size_t)Nn * HEADS * 4;
  unsigned* mkey   = (unsigned*)(ws + o); o += (size_t)Nn * HEADS * 4;  // clear region start
  float*    den    = (float*)(ws + o);    o += (size_t)Nn * HEADS * 4;
  float*    agg    = (float*)(ws + o);    o += (size_t)NPAD * FDIM * 4;
  float*    pooled = (float*)(ws + o);    o += (size_t)Gg * FDIM * 4;   // pooled + cnt contiguous
  float*    cnt    = (float*)(ws + o);    o += (size_t)Gg * 4;

  const int BT = 256;
  const int clearN = Nn * HEADS * 2 + NPAD * FDIM; // mkey + den + agg (contiguous)
  const int poolN  = Gg * FDIM + Gg;               // pooled + cnt (contiguous)

  // ---- once-per-call init ----
  zero_f32<<<cdiv(poolN, BT), BT, 0, stream>>>(pooled, poolN);

  // ================= Layer 1 =================
  cvt_in<<<cdiv((long long)NPAD * KIN, BT), BT, 0, stream>>>(h, x16);
  cvt_wt<<<cdiv(FDIM * KIN, BT), BT, 0, stream>>>(W1, wt16, KIN);
  gemm_wmma<<<NPAD / 32, BT, 0, stream>>>(x16, wt16, feat, KIN);
  attn_lr<<<cdiv(Nn * HEADS, BT), BT, 0, stream>>>(feat, al1, ar1, el, er);
  zero_f32<<<cdiv(clearN, BT), BT, 0, stream>>>((float*)mkey, clearN);
  edge_max<<<cdiv((long long)Ee * HEADS, BT), BT, 0, stream>>>(src, dst, el, er, mkey);
  edge_den<<<cdiv((long long)Ee * HEADS, BT), BT, 0, stream>>>(src, dst, el, er, mkey, den);
  edge_msg<<<cdiv((long long)Ee * HEADS, BT), BT, 0, stream>>>(src, dst, el, er, mkey, den, feat, agg);
  bias_relu_cvt<<<cdiv((long long)NPAD * FDIM, BT), BT, 0, stream>>>(agg, b1, x16);

  // ================= Layer 2 =================
  cvt_wt<<<cdiv(FDIM * FDIM, BT), BT, 0, stream>>>(W2, wt16, FDIM);
  gemm_wmma<<<NPAD / 32, BT, 0, stream>>>(x16, wt16, feat, FDIM);
  attn_lr<<<cdiv(Nn * HEADS, BT), BT, 0, stream>>>(feat, al2, ar2, el, er);
  zero_f32<<<cdiv(clearN, BT), BT, 0, stream>>>((float*)mkey, clearN);
  edge_max<<<cdiv((long long)Ee * HEADS, BT), BT, 0, stream>>>(src, dst, el, er, mkey);
  edge_den<<<cdiv((long long)Ee * HEADS, BT), BT, 0, stream>>>(src, dst, el, er, mkey, den);
  edge_msg<<<cdiv((long long)Ee * HEADS, BT), BT, 0, stream>>>(src, dst, el, er, mkey, den, feat, agg);

  // ================= Pool + MLP head =================
  pool_nodes<<<cdiv((long long)Nn * FDIM, BT), BT, 0, stream>>>(agg, b2, gid, pooled, cnt);
  mlp_head<<<1, BT, 0, stream>>>(pooled, cnt, Wc1, bc1, Wc2, bc2, Wc3, bc3, (float*)d_out);
}
